// NystromAttention_25812753449176
// MI455X (gfx1250) — matrix operand order repaired and checked by reading, not compiled, not run on hardware
//
#include <hip/hip_runtime.h>

typedef _Float16 half_t;
typedef __attribute__((ext_vector_type(16))) half_t v16h;
typedef __attribute__((ext_vector_type(8)))  half_t v8h;
typedef __attribute__((ext_vector_type(8)))  float  v8f;

#define SCALE_QK 0.35355339059327373f   // 1/sqrt(sqrt(64))

__device__ __forceinline__ v8f wmma32(v16h a, v16h b, v8f c) {
    return __builtin_amdgcn_wmma_f32_16x16x32_f16(false, a, false, b, (short)0, c, false, false);
}

__device__ __forceinline__ v8f zero8() {
    v8f z = {0.f, 0.f, 0.f, 0.f, 0.f, 0.f, 0.f, 0.f};
    return z;
}

// A operand: 16x32 f16 tile, rows M = lane&15 from row-major [M][K] memory.
// element e <-> K = 16*(lane>=16) + e
__device__ __forceinline__ v16h load_A(const half_t* tile0, int stride, int k0) {
    const int lane = threadIdx.x & 31;
    const half_t* p = tile0 + (lane & 15) * stride + k0 + ((lane >> 4) << 4);
    v8h lo = *(const v8h*)p;
    v8h hi = *(const v8h*)(p + 8);
    v16h r;
#pragma unroll
    for (int i = 0; i < 8; ++i) { r[i] = lo[i]; r[i + 8] = hi[i]; }
    return r;
}

// B operand: 32x16 f16 tile supplied as row-major [N][K] rows (NT form).
// lane: N = lane&15; element e <-> K = 16*(lane>=16) + e  (contiguous 16 halfs)
__device__ __forceinline__ v16h load_B(const half_t* tile0, int stride, int k0) {
    const int lane = threadIdx.x & 31;
    const half_t* p = tile0 + (lane & 15) * stride + k0 + ((lane >> 4) << 4);
    return *(const v16h*)p;
}

__device__ __forceinline__ v16h load_A_exp(const half_t* tile0, int stride, int k0, float m) {
    v16h raw = load_A(tile0, stride, k0);
    v16h r;
#pragma unroll
    for (int i = 0; i < 16; ++i) r[i] = (half_t)__expf((float)raw[i] - m);
    return r;
}

// monotone float->uint key for atomic max over possibly-negative floats
__device__ __forceinline__ unsigned fkey(float f) {
    unsigned u = __float_as_uint(f);
    return (u & 0x80000000u) ? ~u : (u | 0x80000000u);
}
__device__ __forceinline__ float funkey(unsigned u) {
    return (u & 0x80000000u) ? __uint_as_float(u & 0x7fffffffu) : __uint_as_float(~u);
}

// ---------------- prep: fp32 -> f16 conversions / weight transposes ----------------

__global__ void k_cvt(const float* __restrict__ x, half_t* __restrict__ xh, int n) {
    int i = blockIdx.x * 256 + threadIdx.x;
    if (i < n) xh[i] = (half_t)x[i];
}

// W [K=512][N=512] row-major  ->  Wt [N][K] f16
__global__ void k_twt(const float* __restrict__ w, half_t* __restrict__ wt) {
    int i = blockIdx.x * 256 + threadIdx.x;   // 0..262143
    int n = i >> 9, k = i & 511;
    wt[n * 512 + k] = (half_t)w[k * 512 + n];
}

// ---------------- QKV projection: [32768x512] @ Wqkv^T[1536x512] ----------------
// writes qh,kh [bh][s][d] f16 (scaled), vT [bh][d][s] f16

__global__ __launch_bounds__(128) void k_qkv(
    const half_t* __restrict__ xh, const half_t* __restrict__ wt,
    const float* __restrict__ bq, const float* __restrict__ bk, const float* __restrict__ bv,
    half_t* __restrict__ qh, half_t* __restrict__ kh, half_t* __restrict__ vT) {
    const int lane = threadIdx.x & 31;
    const int wv = threadIdx.x >> 5;
    const int Mbase = blockIdx.x * 256 + wv * 64;   // [0,32768)
    const int nb = blockIdx.y * 64;                 // [0,1536)
    v8f D[4][4];
    for (int mt = 0; mt < 4; ++mt)
        for (int nt = 0; nt < 4; ++nt) D[mt][nt] = zero8();
    for (int kt = 0; kt < 16; ++kt) {
        const int k0 = kt * 32;
        v16h a[4];
#pragma unroll
        for (int mt = 0; mt < 4; ++mt)
            a[mt] = load_A(xh + (Mbase + mt * 16) * 512, 512, k0);
#pragma unroll
        for (int nt = 0; nt < 4; ++nt) {
            v16h b = load_B(wt + (nb + nt * 16) * 512, 512, k0);
#pragma unroll
            for (int mt = 0; mt < 4; ++mt) D[mt][nt] = wmma32(a[mt], b, D[mt][nt]);
        }
    }
    for (int nt = 0; nt < 4; ++nt) {
        const int n = nb + nt * 16 + (lane & 15);
        const int which = n >> 9;       // 0=q 1=k 2=v
        const int hn = n & 511;
        const int h = hn >> 6, d = hn & 63;
        const float bias = (which == 0) ? bq[hn] : (which == 1) ? bk[hn] : bv[hn];
        for (int mt = 0; mt < 4; ++mt) {
#pragma unroll
            for (int j = 0; j < 8; ++j) {
                const int m = Mbase + mt * 16 + j + ((lane >> 4) << 3);
                const int b_ = m >> 13, s = m & 8191;
                const int bh = b_ * 8 + h;
                float v = D[mt][nt][j] + bias;
                if (which < 2) v *= SCALE_QK;
                if (which == 0)      qh[(bh * 8192 + s) * 64 + d] = (half_t)v;
                else if (which == 1) kh[(bh * 8192 + s) * 64 + d] = (half_t)v;
                else                 vT[(bh * 64 + d) * 8192 + s] = (half_t)v;
            }
        }
    }
}

// ---------------- landmark means (seg = 128) ----------------

__global__ void k_land(const half_t* __restrict__ qh, const half_t* __restrict__ kh,
                       half_t* __restrict__ qland, half_t* __restrict__ kland) {
    int idx = blockIdx.x * 256 + threadIdx.x;   // 0..262143
    int which = idx >> 17;
    int r = idx & 131071;
    int bh = r >> 12;
    int ld = r & 4095;
    int l = ld >> 6, d = ld & 63;
    const half_t* src = (which == 0 ? qh : kh) + (bh * 8192 + l * 128) * 64 + d;
    float s = 0.f;
    for (int i = 0; i < 128; ++i) s += (float)src[i * 64];
    (which == 0 ? qland : kland)[bh * 4096 + ld] = (half_t)(s * (1.0f / 128.0f));
}

// ---------------- init: gmax scalar + B3 accumulator ----------------

__global__ void k_init(int* gmax, float* B3) {
    int idx = blockIdx.x * 256 + threadIdx.x;
    if (idx == 0) *gmax = 0;
    if (idx < 32 * 4096) B3[idx] = 0.f;
}

// ---------------- kernel_2 = softmax(q_land @ k_land^T); colsum max ----------------

__global__ __launch_bounds__(128) void k_k2(const half_t* __restrict__ qland,
                                            const half_t* __restrict__ kland,
                                            float* __restrict__ K2, int* gmax) {
    __shared__ __attribute__((aligned(16))) float Sm[64 * 64];
    __shared__ float cs[64];
    const int bh = blockIdx.x;
    const int lane = threadIdx.x & 31, wv = threadIdx.x >> 5;
    const half_t* ql = qland + bh * 4096;
    const half_t* kl = kland + bh * 4096;
    v16h a0 = load_A(ql + wv * 16 * 64, 64, 0);
    v16h a1 = load_A(ql + wv * 16 * 64, 64, 32);
    for (int nt = 0; nt < 4; ++nt) {
        v8f dd = zero8();
        dd = wmma32(a0, load_B(kl + nt * 16 * 64, 64, 0), dd);
        dd = wmma32(a1, load_B(kl + nt * 16 * 64, 64, 32), dd);
#pragma unroll
        for (int j = 0; j < 8; ++j)
            Sm[(wv * 16 + j + ((lane >> 4) << 3)) * 64 + nt * 16 + (lane & 15)] = dd[j];
    }
    __syncthreads();
    const int t = threadIdx.x;
    if (t < 64) {
        float m = -1e30f;
        for (int c = 0; c < 64; ++c) m = fmaxf(m, Sm[t * 64 + c]);
        float s = 0.f;
        for (int c = 0; c < 64; ++c) { float e = __expf(Sm[t * 64 + c] - m); Sm[t * 64 + c] = e; s += e; }
        float inv = 1.f / s;
        for (int c = 0; c < 64; ++c) {
            float v = Sm[t * 64 + c] * inv;
            Sm[t * 64 + c] = v;
            K2[bh * 4096 + t * 64 + c] = v;
        }
    }
    __syncthreads();
    if (t < 64) {
        float s = 0.f;
        for (int r = 0; r < 64; ++r) s += Sm[r * 64 + t];
        cs[t] = s;
    }
    __syncthreads();
    if (t == 0) {
        float m = 0.f;
        for (int c = 0; c < 64; ++c) m = fmaxf(m, cs[c]);
        atomicMax(gmax, __float_as_int(m));   // positive floats order as ints
    }
}

// ---------------- kernel_3 logits: Lg[l][s] = q_land[l].k[s]; rowmax/rowsum ----------------
// computed as (k @ q_land^T) so D tiles store contiguously along s

__global__ __launch_bounds__(128) void k_k3a(const half_t* __restrict__ kh,
                                             const half_t* __restrict__ qland,
                                             half_t* __restrict__ Lg,
                                             float* __restrict__ rmax, float* __restrict__ rsum) {
    __shared__ unsigned rk[64];
    __shared__ float rs[64], rm[64];
    const int bh = blockIdx.x;
    const int lane = threadIdx.x & 31, wv = threadIdx.x >> 5;
    const half_t* kbh = kh + (size_t)bh * 8192 * 64;
    const half_t* ql = qland + bh * 4096;
    half_t* Lbh = Lg + (size_t)bh * 64 * 8192;
    if (threadIdx.x < 64) { rk[threadIdx.x] = 0u; rs[threadIdx.x] = 0.f; }
    __syncthreads();
    v16h B[4][2];
#pragma unroll
    for (int nt = 0; nt < 4; ++nt) {
        B[nt][0] = load_B(ql + nt * 16 * 64, 64, 0);
        B[nt][1] = load_B(ql + nt * 16 * 64, 64, 32);
    }
    for (int ch = 0; ch < 128; ++ch) {
        const int st = ch * 64 + wv * 16;
        v16h a0 = load_A(kbh + st * 64, 64, 0);
        v16h a1 = load_A(kbh + st * 64, 64, 32);
        for (int nt = 0; nt < 4; ++nt) {
            v8f dd = zero8();
            dd = wmma32(a0, B[nt][0], dd);
            dd = wmma32(a1, B[nt][1], dd);
            const int l = nt * 16 + (lane & 15);
            half_t* dst = Lbh + l * 8192 + st + ((lane >> 4) << 3);
            float mloc = -1e30f;
            v8h hv;
#pragma unroll
            for (int j = 0; j < 8; ++j) { float v = dd[j]; mloc = fmaxf(mloc, v); hv[j] = (half_t)v; }
            *(v8h*)dst = hv;
            atomicMax(&rk[l], fkey(mloc));
        }
    }
    __syncthreads();
    if (threadIdx.x < 64) {
        float m = funkey(rk[threadIdx.x]);
        rm[threadIdx.x] = m;
        rmax[bh * 64 + threadIdx.x] = m;
    }
    __syncthreads();
    {
        const int t = threadIdx.x;
        const int l = t & 63, hf = t >> 6;
        const half_t* row = Lbh + l * 8192 + hf * 4096;
        float m = rm[l], p = 0.f;
        for (int s = 0; s < 4096; ++s) p += __expf((float)row[s] - m);
        atomicAdd(&rs[l], p);
    }
    __syncthreads();
    if (threadIdx.x < 64) rsum[bh * 64 + threadIdx.x] = rs[threadIdx.x];
}

// ---------------- B3 = softmax(kernel_3) @ v  (split-K over s, atomic accumulate) ----------------

__global__ __launch_bounds__(128) void k_k3b(const half_t* __restrict__ Lg,
                                             const half_t* __restrict__ vT,
                                             const float* __restrict__ rmax,
                                             const float* __restrict__ rsum,
                                             float* __restrict__ B3) {
    __shared__ float rm[64], rsl[64];
    const int bh = blockIdx.x;
    const int ks = blockIdx.y;   // 0..7
    const int lane = threadIdx.x & 31, wv = threadIdx.x >> 5;
    if (threadIdx.x < 64) {
        rm[threadIdx.x] = rmax[bh * 64 + threadIdx.x];
        rsl[threadIdx.x] = rsum[bh * 64 + threadIdx.x];
    }
    __syncthreads();
    const half_t* Lrow0 = Lg + (size_t)bh * 64 * 8192 + wv * 16 * 8192;
    const half_t* vbh = vT + (size_t)bh * 64 * 8192;
    const float mlane = rm[wv * 16 + (lane & 15)];
    v8f D4[4];
    for (int nt = 0; nt < 4; ++nt) D4[nt] = zero8();
    for (int kt = 0; kt < 32; ++kt) {
        const int k0 = ks * 1024 + kt * 32;
        v16h a = load_A_exp(Lrow0, 8192, k0, mlane);
#pragma unroll
        for (int nt = 0; nt < 4; ++nt) {
            v16h b = load_B(vbh + nt * 16 * 8192, 8192, k0);
            D4[nt] = wmma32(a, b, D4[nt]);
        }
    }
    for (int nt = 0; nt < 4; ++nt) {
        const int d = nt * 16 + (lane & 15);
#pragma unroll
        for (int j = 0; j < 8; ++j) {
            const int l = wv * 16 + j + ((lane >> 4) << 3);
            atomicAdd(&B3[bh * 4096 + l * 64 + d], D4[nt][j] / rsl[l]);
        }
    }
}

// ---------------- Newton-Schulz pseudo-inverse + C2^T = (inv @ B3)^T ----------------

__device__ __forceinline__ void mm64(const float* A, const float* B, float* C, int t) {
    for (int i = 0; i < 16; ++i) {
        int o = t + (i << 8);
        int r = o >> 6, c = o & 63;
        float s = 0.f;
        for (int kk = 0; kk < 64; ++kk) s += A[r * 64 + kk] * B[kk * 64 + c];
        C[o] = s;
    }
}
__device__ __forceinline__ void ew_isub(float a, const float* M, float* O, int t) {
    for (int i = 0; i < 16; ++i) {
        int o = t + (i << 8);
        int r = o >> 6, c = o & 63;
        O[o] = (r == c ? a : 0.0f) - M[o];
    }
}

__global__ __launch_bounds__(256) void k_newton(const float* __restrict__ K2, const int* gmax,
                                                const float* __restrict__ B3,
                                                half_t* __restrict__ C2T) {
    __shared__ float Vm[4096], KV[4096], T1[4096], T2[4096];
    const int bh = blockIdx.x;
    const int t = threadIdx.x;
    const float ginv = 1.0f / __int_as_float(*gmax);
    const float* K2g = K2 + bh * 4096;
    for (int i = 0; i < 16; ++i) {
        int o = t + (i << 8);
        int r = o >> 6, c = o & 63;
        Vm[o] = K2g[c * 64 + r] * ginv;   // V = K^T / gmax
    }
    __syncthreads();
    for (int it = 0; it < 6; ++it) {
        mm64(K2g, Vm, KV, t); __syncthreads();
        ew_isub(7.f, KV, T1, t); __syncthreads();
        mm64(KV, T1, T2, t); __syncthreads();
        ew_isub(15.f, T2, T1, t); __syncthreads();
        mm64(KV, T1, T2, t); __syncthreads();
        ew_isub(13.f, T2, T1, t); __syncthreads();
        mm64(Vm, T1, T2, t); __syncthreads();
        for (int i = 0; i < 16; ++i) { int o = t + (i << 8); Vm[o] = 0.25f * T2[o]; }
        __syncthreads();
    }
    for (int i = 0; i < 16; ++i) { int o = t + (i << 8); T1[o] = B3[bh * 4096 + o]; }
    __syncthreads();
    mm64(Vm, T1, T2, t); __syncthreads();
    for (int i = 0; i < 16; ++i) {
        int o = t + (i << 8);
        int r = o >> 6, c = o & 63;
        C2T[bh * 4096 + c * 64 + r] = (half_t)T2[o];
    }
}

// ---------------- epilogue: out_heads = softmax(q @ k_land^T) @ C2 ----------------

__global__ __launch_bounds__(128) void k_epi(const half_t* __restrict__ qh,
                                             const half_t* __restrict__ kland,
                                             const half_t* __restrict__ C2T,
                                             half_t* __restrict__ OH) {
    __shared__ __attribute__((aligned(16))) float Pbuf[64 * 64];
    __shared__ __attribute__((aligned(16))) half_t Ph[64 * 64];
    const int bh = blockIdx.x >> 7;
    const int sc = blockIdx.x & 127;
    const int lane = threadIdx.x & 31, wv = threadIdx.x >> 5;
    const int s0 = sc * 64 + wv * 16;
    const half_t* qrow = qh + ((size_t)bh * 8192 + s0) * 64;
    const half_t* kl = kland + bh * 4096;
    const half_t* c2 = C2T + bh * 4096;
    v16h a0 = load_A(qrow, 64, 0);
    v16h a1 = load_A(qrow, 64, 32);
    for (int nt = 0; nt < 4; ++nt) {
        v8f dd = zero8();
        dd = wmma32(a0, load_B(kl + nt * 16 * 64, 64, 0), dd);
        dd = wmma32(a1, load_B(kl + nt * 16 * 64, 64, 32), dd);
        const int l = nt * 16 + (lane & 15);
#pragma unroll
        for (int j = 0; j < 8; ++j)
            Pbuf[(wv * 16 + j + ((lane >> 4) << 3)) * 64 + l] = dd[j];
    }
    __syncthreads();
    if (threadIdx.x < 64) {
        const int r = threadIdx.x;
        float m = -1e30f;
        for (int c = 0; c < 64; ++c) m = fmaxf(m, Pbuf[r * 64 + c]);
        float s = 0.f;
        for (int c = 0; c < 64; ++c) { float e = __expf(Pbuf[r * 64 + c] - m); Pbuf[r * 64 + c] = e; s += e; }
        float inv = 1.f / s;
        for (int c = 0; c < 64; ++c) Ph[r * 64 + c] = (half_t)(Pbuf[r * 64 + c] * inv);
    }
    __syncthreads();
    v16h p0 = load_A(Ph + wv * 16 * 64, 64, 0);
    v16h p1 = load_A(Ph + wv * 16 * 64, 64, 32);
    const int b_ = bh >> 3, h = bh & 7;
    for (int nt = 0; nt < 4; ++nt) {
        v8f dd = zero8();
        dd = wmma32(p0, load_B(c2 + nt * 16 * 64, 64, 0), dd);
        dd = wmma32(p1, load_B(c2 + nt * 16 * 64, 64, 32), dd);
        const int d = nt * 16 + (lane & 15);
#pragma unroll
        for (int j = 0; j < 8; ++j) {
            const int s = s0 + j + ((lane >> 4) << 3);
            OH[(size_t)(b_ * 8192 + s) * 512 + h * 64 + d] = (half_t)dd[j];
        }
    }
}

// ---------------- output projection: out = OH @ Wo^T + bo (f32) ----------------

__global__ __launch_bounds__(128) void k_out(const half_t* __restrict__ OH,
                                             const half_t* __restrict__ wot,
                                             const float* __restrict__ bo,
                                             float* __restrict__ out) {
    const int lane = threadIdx.x & 31;
    const int wv = threadIdx.x >> 5;
    const int Mbase = blockIdx.x * 256 + wv * 64;
    const int nb = blockIdx.y * 64;
    v8f D[4][4];
    for (int mt = 0; mt < 4; ++mt)
        for (int nt = 0; nt < 4; ++nt) D[mt][nt] = zero8();
    for (int kt = 0; kt < 16; ++kt) {
        const int k0 = kt * 32;
        v16h a[4];
#pragma unroll
        for (int mt = 0; mt < 4; ++mt)
            a[mt] = load_A(OH + (size_t)(Mbase + mt * 16) * 512, 512, k0);
#pragma unroll
        for (int nt = 0; nt < 4; ++nt) {
            v16h b = load_B(wot + (nb + nt * 16) * 512, 512, k0);
#pragma unroll
            for (int mt = 0; mt < 4; ++mt) D[mt][nt] = wmma32(a[mt], b, D[mt][nt]);
        }
    }
    for (int nt = 0; nt < 4; ++nt) {
        const int e = nb + nt * 16 + (lane & 15);
        const float bias = bo[e];
        for (int mt = 0; mt < 4; ++mt) {
#pragma unroll
            for (int j = 0; j < 8; ++j) {
                const int m = Mbase + mt * 16 + j + ((lane >> 4) << 3);
                out[(size_t)m * 512 + e] = D[mt][nt][j] + bias;
            }
        }
    }
}

// ---------------- host launcher ----------------

extern "C" void kernel_launch(void* const* d_in, const int* in_sizes, int n_in,
                              void* d_out, int out_size, void* d_ws, size_t ws_size,
                              hipStream_t stream) {
    (void)in_sizes; (void)n_in; (void)out_size; (void)ws_size;
    const float* x  = (const float*)d_in[0];
    const float* Wq = (const float*)d_in[1];
    const float* bq = (const float*)d_in[2];
    const float* Wk = (const float*)d_in[3];
    const float* bk = (const float*)d_in[4];
    const float* Wv = (const float*)d_in[5];
    const float* bv = (const float*)d_in[6];
    const float* Wo = (const float*)d_in[7];
    const float* bo = (const float*)d_in[8];
    float* out = (float*)d_out;

    char* ws = (char*)d_ws;
    auto carve = [&](size_t bytes) -> void* {
        void* p = (void*)ws;
        ws += (bytes + 255) & ~(size_t)255;
        return p;
    };
    const size_t NX = 16777216;   // 4*8192*512
    half_t* xh    = (half_t*)carve(NX * 2);
    half_t* wqkvt = (half_t*)carve(3 * 262144 * 2);
    half_t* wot   = (half_t*)carve(262144 * 2);
    half_t* qh    = (half_t*)carve(NX * 2);
    half_t* kh    = (half_t*)carve(NX * 2);
    half_t* vT    = (half_t*)carve(NX * 2);
    half_t* qland = (half_t*)carve(131072 * 2);
    half_t* kland = (half_t*)carve(131072 * 2);
    float*  K2    = (float*) carve(131072 * 4);
    float*  B3    = (float*) carve(131072 * 4);
    half_t* C2T   = (half_t*)carve(131072 * 2);
    half_t* Lg    = (half_t*)carve(NX * 2);
    float*  rmax  = (float*) carve(2048 * 4);
    float*  rsum  = (float*) carve(2048 * 4);
    half_t* OH    = (half_t*)carve(NX * 2);
    int*    gmax  = (int*)   carve(256);

    k_cvt<<<65536, 256, 0, stream>>>(x, xh, (int)NX);
    k_twt<<<1024, 256, 0, stream>>>(Wq, wqkvt);
    k_twt<<<1024, 256, 0, stream>>>(Wk, wqkvt + 262144);
    k_twt<<<1024, 256, 0, stream>>>(Wv, wqkvt + 2 * 262144);
    k_twt<<<1024, 256, 0, stream>>>(Wo, wot);
    k_qkv<<<dim3(128, 24), 128, 0, stream>>>(xh, wqkvt, bq, bk, bv, qh, kh, vT);
    k_land<<<1024, 256, 0, stream>>>(qh, kh, qland, kland);
    k_init<<<512, 256, 0, stream>>>(gmax, B3);
    k_k2<<<32, 128, 0, stream>>>(qland, kland, K2, gmax);
    k_k3a<<<32, 128, 0, stream>>>(kh, qland, Lg, rmax, rsum);
    k_k3b<<<dim3(32, 8), 128, 0, stream>>>(Lg, vT, rmax, rsum, B3);
    k_newton<<<32, 256, 0, stream>>>(K2, gmax, B3, C2T);
    k_epi<<<4096, 128, 0, stream>>>(qh, kland, C2T, OH);
    k_out<<<dim3(128, 8), 128, 0, stream>>>(OH, wot, bo, out);
}
